// GaussianRefineHeadSparseConv_40750649704651
// MI455X (gfx1250) — compile-verified, matrix operand order checked by reading, not run on hardware
//
#include <hip/hip_runtime.h>
#include <hip/hip_bf16.h>

// ---------------- CDNA5 WMMA / TDM types ----------------
typedef __attribute__((ext_vector_type(16))) __bf16 v16bf;
typedef __attribute__((ext_vector_type(8)))  float  v8f;
typedef __attribute__((ext_vector_type(4)))  unsigned int u32x4;
typedef __attribute__((ext_vector_type(8)))  int i32x8;
typedef __attribute__((ext_vector_type(4)))  int i32x4;

#define CCH 128
#define EPSN 1e-5f

__device__ __forceinline__ unsigned short f2bf(float f) {
  unsigned u = __float_as_uint(f);
  unsigned r = u + 0x7FFFu + ((u >> 16) & 1u);   // round-to-nearest-even
  return (unsigned short)(r >> 16);
}

union FragAB { v16bf v; uint4 q[2]; };

// Issue one TDM DMA: 32KB contiguous (4096 x 8B) global -> LDS.
// D# group0: count=1 | lds_addr | global_addr[56:0] | type=2
// D# group1: data_size=8B, tensor_dim0=tile_dim0=stride0=4096, dim1=1
__device__ __forceinline__ void tdm_stage32k(const unsigned short* gsrc, unsigned lds_lo) {
  unsigned long long ga = (unsigned long long)gsrc;
  u32x4 g0;
  g0[0] = 1u;                                               // count=1, is_restore=0
  g0[1] = lds_lo;                                           // lds_addr (bytes)
  g0[2] = (unsigned)ga;                                     // global_addr[31:0]
  g0[3] = ((unsigned)(ga >> 32) & 0x01FFFFFFu) | (2u << 30); // ga[56:32] | type=2
  i32x8 g1;
  g1[0] = (3 << 16);            // workgroup_mask=0, data_size=3 (8 bytes)
  g1[1] = (int)(4096u << 16);   // tensor_dim0[15:0] @ bits 63:48
  g1[2] = (1 << 16);            // tensor_dim0[31:16]=0, tensor_dim1=1 @ bits 111:80
  g1[3] = (int)(4096u << 16);   // tile_dim0=4096 @ bits 127:112
  g1[4] = 1;                    // tile_dim1=1, tile_dim2=0
  g1[5] = 4096;                 // tensor_dim0_stride[31:0]
  g1[6] = 0;                    // stride high bits / dim1_stride
  g1[7] = 0;
  i32x4 gz4 = {0, 0, 0, 0};
  i32x8 gz8 = {0, 0, 0, 0, 0, 0, 0, 0};
  __builtin_amdgcn_tensor_load_to_lds(g0, g1, gz4, gz4, gz8, 0);
}

// ---------------- zero init ----------------
__global__ void zero_f32(float* p, size_t n) {
  size_t i = (size_t)blockIdx.x * blockDim.x + threadIdx.x;
  size_t st = (size_t)gridDim.x * blockDim.x;
  for (; i < n; i += st) p[i] = 0.f;
}

// ---------------- pack a KxNsrc f32 matrix into per-lane WMMA B-fragment bf16 layout
// packed element index (per matrix): ((ks*NT + nt)*32 + lane)*16 + j
__global__ void __launch_bounds__(256)
pack_b(const float* __restrict__ src, unsigned short* __restrict__ dst,
       int K, int Nsrc, int NT, int KS) {
  int mat = blockIdx.x;
  const float* s = src + (size_t)mat * K * Nsrc;
  unsigned short* d = dst + (size_t)mat * NT * KS * 512;
  int total = NT * KS * 512;
  for (int idx = threadIdx.x; idx < total; idx += blockDim.x) {
    int j    = idx & 15;
    int lane = (idx >> 4) & 31;
    int q    = idx >> 9;          // = ks*NT + nt
    int nt   = q % NT;
    int ks   = q / NT;
    int n    = nt * 16 + (lane & 15);
    int kb   = (lane >> 4) * 8;
    int k    = ks * 32 + kb + ((j < 8) ? j : (16 + j - 8));
    float val = (k < K && n < Nsrc) ? s[(size_t)k * Nsrc + n] : 0.f;
    d[idx] = f2bf(val);
  }
}

// ---------------- encoder: Linear(14->128) + LayerNorm + ReLU + atomic segment-sum
__global__ void __launch_bounds__(256)
encoder_kernel(const float* __restrict__ gp, const float* __restrict__ wenc,
               const float* __restrict__ benc, const float* __restrict__ lng,
               const float* __restrict__ lnb, const int* __restrict__ inv_idx,
               float* __restrict__ xsum, float* __restrict__ cnt, int Npts) {
  int wave = threadIdx.x >> 5, lane = threadIdx.x & 31;
  int p = blockIdx.x * 8 + wave;
  if (p >= Npts) return;
  float g[14];
#pragma unroll
  for (int k = 0; k < 14; ++k) g[k] = gp[(size_t)p * 14 + k];
  float f[4];
#pragma unroll
  for (int j = 0; j < 4; ++j) {
    int c = lane + j * 32;
    float a = benc[c];
#pragma unroll
    for (int k = 0; k < 14; ++k) a += g[k] * wenc[k * CCH + c];
    f[j] = a;
  }
  float s1 = f[0] + f[1] + f[2] + f[3];
  float s2 = f[0]*f[0] + f[1]*f[1] + f[2]*f[2] + f[3]*f[3];
#pragma unroll
  for (int d = 16; d >= 1; d >>= 1) {           // wave32 butterfly
    s1 += __shfl_xor(s1, d, 32);
    s2 += __shfl_xor(s2, d, 32);
  }
  float mean = s1 * (1.f / 128.f);
  float var  = s2 * (1.f / 128.f) - mean * mean;
  float rs   = rsqrtf(var + EPSN);
  int v = inv_idx[p];
#pragma unroll
  for (int j = 0; j < 4; ++j) {
    int c = lane + j * 32;
    float val = (f[j] - mean) * rs * lng[c] + lnb[c];
    val = val > 0.f ? val : 0.f;
    atomicAdd(&xsum[(size_t)v * CCH + c], val);
  }
  if (lane == 0) atomicAdd(&cnt[v], 1.0f);
}

// ---------------- voxel mean + bf16 shadow --------------------------------------
__global__ void __launch_bounds__(256)
vox_finalize(float* __restrict__ xsum, const float* __restrict__ cnt,
             unsigned short* __restrict__ xbf, const int* __restrict__ Mptr) {
  int M = *Mptr;
  size_t i = (size_t)blockIdx.x * 256 + threadIdx.x;
  if (i >= (size_t)M * CCH) return;
  int r = (int)(i >> 7);
  float val = xsum[i] / cnt[r];
  xsum[i] = val;
  xbf[i]  = f2bf(val);
}

// ---------------- submanifold 3x3x3 conv via WMMA bf16 --------------------------
// block = 4 waves; wave w handles rows [ (blockIdx*4+w)*16 , +16 ).
// W[o] staged to LDS by the Tensor Data Mover (wave0 issues; TENSORcnt tracks),
// double-buffered so the DMA of W[o+1] overlaps the 32 WMMAs of W[o].
__global__ void __launch_bounds__(128)
conv_subm_wmma(const unsigned short* __restrict__ xb,
               const int*   __restrict__ nbr_idx,
               const float* __restrict__ nbr_mask,
               const unsigned short* __restrict__ wpack,  // 27 mats * 16384 bf16
               const int* __restrict__ Mptr,
               float* __restrict__ y) {
  __shared__ __align__(16) unsigned short sW[2][16384];   // 2 x 32 KB
  const int M    = *Mptr;
  const int tid  = threadIdx.x;
  const int wave = tid >> 5;
  const int lane = tid & 31;
  const int mrow = lane & 15;
  const int khalf = (lane >> 4) * 8;
  const int base = (blockIdx.x * 4 + wave) * 16;
  const int row  = base + mrow;
  const bool rowok = row < M;

  const unsigned lds0 = (unsigned)(unsigned long long)(void*)&sW[0][0];
  const unsigned lds1 = (unsigned)(unsigned long long)(void*)&sW[1][0];

  v8f acc[8];
#pragma unroll
  for (int i = 0; i < 8; ++i) acc[i] = v8f{0.f,0.f,0.f,0.f,0.f,0.f,0.f,0.f};

  const uint4 zero4 = make_uint4(0u, 0u, 0u, 0u);

  if (wave == 0)                              // uniform branch: one TDM per block
    tdm_stage32k(wpack, lds0);

  for (int o = 0; o < 27; ++o) {
    const int b = o & 1;
    if (wave == 0)
      __builtin_amdgcn_s_wait_tensorcnt((short)0);   // W[o] landed in LDS
    __syncthreads();                                 // visible to all 4 waves
    if (wave == 0 && o + 1 < 27)                     // overlap DMA of W[o+1]
      tdm_stage32k(wpack + (size_t)(o + 1) * 16384, b ? lds0 : lds1);

    int nidx = 0; float mval = 0.f;
    if (rowok) { nidx = nbr_idx[row * 27 + o]; mval = nbr_mask[row * 27 + o]; }
    const bool use = rowok && (mval > 0.5f);
    const uint4* arow = (const uint4*)(xb + (size_t)nidx * CCH);
    const unsigned short* wb = &sW[b][0];

#pragma unroll
    for (int ks = 0; ks < 4; ++ks) {
      FragAB af;
      if (use) {
        af.q[0] = arow[(ks * 32 + khalf) >> 3];
        af.q[1] = arow[(ks * 32 + khalf + 16) >> 3];
      } else { af.q[0] = zero4; af.q[1] = zero4; }
#pragma unroll
      for (int nt = 0; nt < 8; ++nt) {
        FragAB bf;
        const uint4* bs = (const uint4*)(wb + (size_t)((ks * 8 + nt) * 32 + lane) * 16);
        bf.q[0] = bs[0];
        bf.q[1] = bs[1];
        acc[nt] = __builtin_amdgcn_wmma_f32_16x16x32_bf16(
            false, af.v, false, bf.v, (short)0, acc[nt], false, false);
      }
    }
  }
  // C/D layout: m = v + 8*(lane>=16), n = nt*16 + (lane&15)
#pragma unroll
  for (int nt = 0; nt < 8; ++nt) {
#pragma unroll
    for (int v = 0; v < 8; ++v) {
      int r = base + v + ((lane >> 4) << 3);
      if (r < M) y[(size_t)r * CCH + nt * 16 + (lane & 15)] = acc[nt][v];
    }
  }
}

// ---------------- BN (training-mode batch stats) --------------------------------
__global__ void __launch_bounds__(128)
bn_stats(const float* __restrict__ y, const int* __restrict__ Mptr,
         float* __restrict__ s, float* __restrict__ ss) {
  int M = *Mptr; int c = threadIdx.x;
  float a = 0.f, b = 0.f;
  for (int r = blockIdx.x; r < M; r += gridDim.x) {
    float v = y[(size_t)r * CCH + c];
    a += v; b += v * v;
  }
  atomicAdd(&s[c], a);
  atomicAdd(&ss[c], b);
}

__global__ void bn_finalize(const float* __restrict__ s, const float* __restrict__ ss,
                            const float* __restrict__ g, const float* __restrict__ b,
                            const int* __restrict__ Mptr,
                            float* __restrict__ scale, float* __restrict__ shift) {
  int M = *Mptr; int c = threadIdx.x;
  float inv = 1.0f / (float)M;
  float mean = s[c] * inv;
  float var  = ss[c] * inv - mean * mean;
  float sc   = g[c] * rsqrtf(var + EPSN);
  scale[c] = sc;
  shift[c] = b[c] - mean * sc;
}

__global__ void __launch_bounds__(256)
apply_bn_relu(const float* __restrict__ y, const float* __restrict__ scale,
              const float* __restrict__ shift, const int* __restrict__ Mptr,
              unsigned short* __restrict__ tbf) {
  int M = *Mptr;
  size_t i = (size_t)blockIdx.x * 256 + threadIdx.x;
  if (i >= (size_t)M * CCH) return;
  int c = (int)(i & 127);
  float v = y[i] * scale[c] + shift[c];
  v = v > 0.f ? v : 0.f;
  tbf[i] = f2bf(v);
}

__global__ void __launch_bounds__(256)
apply_bn_relu_res(const float* __restrict__ y, const float* __restrict__ scale,
                  const float* __restrict__ shift, const int* __restrict__ Mptr,
                  float* __restrict__ xf, unsigned short* __restrict__ xbf) {
  int M = *Mptr;
  size_t i = (size_t)blockIdx.x * 256 + threadIdx.x;
  if (i >= (size_t)M * CCH) return;
  int c = (int)(i & 127);
  float v = y[i] * scale[c] + shift[c] + xf[i];
  v = v > 0.f ? v : 0.f;
  xf[i]  = v;
  xbf[i] = f2bf(v);
}

// ---------------- head: gather + (128->64 relu) @ (64->14) via WMMA -------------
__global__ void __launch_bounds__(128)
head_wmma(const unsigned short* __restrict__ xb, const int* __restrict__ inv_idx,
          const unsigned short* __restrict__ w1p,   // NT=4, KS=4 packed
          const float* __restrict__ b1,
          const unsigned short* __restrict__ w2p,   // NT=1, KS=2 packed (padded N->16)
          const float* __restrict__ b2,
          const float* __restrict__ gp, float* __restrict__ out, int Npts) {
  __shared__ __align__(16) unsigned short hbuf[4 * 16 * 64];  // per-wave 16x64 bf16
  int tid = threadIdx.x, wave = tid >> 5, lane = tid & 31;
  int mrow = lane & 15, khalf = (lane >> 4) * 8;
  int base = (blockIdx.x * 4 + wave) * 16;
  int p = base + mrow; if (p >= Npts) p = Npts - 1;
  int vrow = inv_idx[p];
  const uint4* arow = (const uint4*)(xb + (size_t)vrow * CCH);

  v8f h[4];
#pragma unroll
  for (int i = 0; i < 4; ++i) h[i] = v8f{0.f,0.f,0.f,0.f,0.f,0.f,0.f,0.f};

#pragma unroll
  for (int ks = 0; ks < 4; ++ks) {
    FragAB af;
    af.q[0] = arow[(ks * 32 + khalf) >> 3];
    af.q[1] = arow[(ks * 32 + khalf + 16) >> 3];
#pragma unroll
    for (int nt = 0; nt < 4; ++nt) {
      FragAB bf;
      const uint4* bs = (const uint4*)(w1p + (size_t)((ks * 4 + nt) * 32 + lane) * 16);
      bf.q[0] = bs[0]; bf.q[1] = bs[1];
      h[nt] = __builtin_amdgcn_wmma_f32_16x16x32_bf16(
          false, af.v, false, bf.v, (short)0, h[nt], false, false);
    }
  }
  // bias + ReLU, transpose C-layout -> row-major bf16 tile in LDS
  unsigned short* myh = hbuf + wave * 1024;
#pragma unroll
  for (int nt = 0; nt < 4; ++nt) {
    int col = nt * 16 + (lane & 15);
    float bias = b1[col];
#pragma unroll
    for (int v = 0; v < 8; ++v) {
      int m = v + ((lane >> 4) << 3);
      float t = h[nt][v] + bias;
      t = t > 0.f ? t : 0.f;
      myh[m * 64 + col] = f2bf(t);
    }
  }
  __syncthreads();

  v8f acc2 = v8f{0.f,0.f,0.f,0.f,0.f,0.f,0.f,0.f};
#pragma unroll
  for (int ks = 0; ks < 2; ++ks) {
    FragAB af;
    af.q[0] = *(const uint4*)(myh + mrow * 64 + ks * 32 + khalf);
    af.q[1] = *(const uint4*)(myh + mrow * 64 + ks * 32 + khalf + 16);
    FragAB bf;
    const uint4* bs = (const uint4*)(w2p + (size_t)(ks * 32 + lane) * 16);
    bf.q[0] = bs[0]; bf.q[1] = bs[1];
    acc2 = __builtin_amdgcn_wmma_f32_16x16x32_bf16(
        false, af.v, false, bf.v, (short)0, acc2, false, false);
  }
#pragma unroll
  for (int v = 0; v < 8; ++v) {
    int r = base + v + ((lane >> 4) << 3);
    int col = lane & 15;
    if (r < Npts && col < 14)
      out[(size_t)r * 14 + col] = gp[(size_t)r * 14 + col] + acc2[v] + b2[col];
  }
}

// ---------------- host side ------------------------------------------------------
extern "C" void kernel_launch(void* const* d_in, const int* in_sizes, int n_in,
                              void* d_out, int out_size, void* d_ws, size_t ws_size,
                              hipStream_t stream) {
  const float* gp    = (const float*)d_in[0];
  const float* wenc  = (const float*)d_in[1];
  const float* benc  = (const float*)d_in[2];
  const float* lng   = (const float*)d_in[3];
  const float* lnb   = (const float*)d_in[4];
  const float* convw = (const float*)d_in[5];
  const float* bng   = (const float*)d_in[6];
  const float* bnb   = (const float*)d_in[7];
  const float* w1    = (const float*)d_in[8];
  const float* b1    = (const float*)d_in[9];
  const float* w2    = (const float*)d_in[10];
  const float* b2    = (const float*)d_in[11];
  const float* nbrm  = (const float*)d_in[12];
  const int*   invi  = (const int*)d_in[13];
  const int*   nbri  = (const int*)d_in[14];
  const int*   Mptr  = (const int*)d_in[15];

  const int N = in_sizes[0] / 14;   // 200000

  // workspace carve (256B aligned regions)
  char* w = (char*)d_ws;
  size_t off = 0;
  auto carve = [&](size_t bytes) -> char* {
    char* p = w + off;
    off = (off + bytes + 255) & ~(size_t)255;
    return p;
  };
  float* xsum = (float*)carve((size_t)N * CCH * 4);   // becomes x_f32 residual
  float* cnt  = (float*)carve((size_t)N * 4);
  float* bns  = (float*)carve(4 * CCH * 4);
  float* bnss = (float*)carve(4 * CCH * 4);
  size_t zbytes = off;                                 // region to zero
  unsigned short* xbf = (unsigned short*)carve((size_t)N * CCH * 2);
  unsigned short* tbf = (unsigned short*)carve((size_t)N * CCH * 2);
  float* ybuf = (float*)carve((size_t)N * CCH * 4);
  unsigned short* wcp = (unsigned short*)carve((size_t)108 * 16384 * 2);
  unsigned short* w1p = (unsigned short*)carve((size_t)8192 * 2);
  unsigned short* w2p = (unsigned short*)carve((size_t)1024 * 2);
  float* bnscale = (float*)carve(4 * CCH * 4);
  float* bnshift = (float*)carve(4 * CCH * 4);
  (void)ws_size; (void)n_in; (void)out_size;

  zero_f32<<<2048, 256, 0, stream>>>((float*)d_ws, zbytes / 4);

  // pack weights into bf16 B-fragment layout
  pack_b<<<108, 256, 0, stream>>>(convw, wcp, 128, 128, 8, 4);
  pack_b<<<1,   256, 0, stream>>>(w1,    w1p, 128,  64, 4, 4);
  pack_b<<<1,   256, 0, stream>>>(w2,    w2p,  64,  14, 1, 2);

  // encoder + segment mean
  encoder_kernel<<<(N + 7) / 8, 256, 0, stream>>>(gp, wenc, benc, lng, lnb, invi,
                                                  xsum, cnt, N);
  unsigned gridE = (unsigned)(((size_t)N * CCH + 255) / 256);
  vox_finalize<<<gridE, 256, 0, stream>>>(xsum, cnt, xbf, Mptr);

  unsigned gridC = (unsigned)((N + 63) / 64);
  for (int blk = 0; blk < 2; ++blk) {
    int l0 = 2 * blk, l1 = 2 * blk + 1;
    conv_subm_wmma<<<gridC, 128, 0, stream>>>(xbf, nbri, nbrm,
        wcp + (size_t)l0 * 27 * 16384, Mptr, ybuf);
    bn_stats<<<512, 128, 0, stream>>>(ybuf, Mptr, bns + l0 * CCH, bnss + l0 * CCH);
    bn_finalize<<<1, 128, 0, stream>>>(bns + l0 * CCH, bnss + l0 * CCH,
        bng + l0 * CCH, bnb + l0 * CCH, Mptr, bnscale + l0 * CCH, bnshift + l0 * CCH);
    apply_bn_relu<<<gridE, 256, 0, stream>>>(ybuf, bnscale + l0 * CCH,
        bnshift + l0 * CCH, Mptr, tbf);

    conv_subm_wmma<<<gridC, 128, 0, stream>>>(tbf, nbri, nbrm,
        wcp + (size_t)l1 * 27 * 16384, Mptr, ybuf);
    bn_stats<<<512, 128, 0, stream>>>(ybuf, Mptr, bns + l1 * CCH, bnss + l1 * CCH);
    bn_finalize<<<1, 128, 0, stream>>>(bns + l1 * CCH, bnss + l1 * CCH,
        bng + l1 * CCH, bnb + l1 * CCH, Mptr, bnscale + l1 * CCH, bnshift + l1 * CCH);
    apply_bn_relu_res<<<gridE, 256, 0, stream>>>(ybuf, bnscale + l1 * CCH,
        bnshift + l1 * CCH, Mptr, xsum, xbf);
  }

  head_wmma<<<gridC, 128, 0, stream>>>(xbf, invi, w1p, b1, w2p, b2, gp,
                                       (float*)d_out, N);
}
